// Gatv2Predictor_80041010528625
// MI455X (gfx1250) — matrix operand order repaired and checked by reading, not compiled, not run on hardware
//
#include <hip/hip_runtime.h>
#include <cstdint>
#include <cstddef>

// ---------------------------------------------------------------------------
// GATv2 transformer predictor forward pass for MI455X (gfx1250, wave32).
// All K=64 GEMMs run on V_WMMA_F32_16X16X32_BF16 (bf16 in, fp32 accumulate).
// Each wave computes a 16x64 output strip: A fragments built once, reused
// across four B tiles -> 8 WMMAs per staged A tile.
// Decoder self/cross attention has S=T=1 -> softmax==1 -> reduces to V,O
// projections only (Q/K projections and the g2 LayerNorm are dead code).
// ---------------------------------------------------------------------------

#define HID   64
#define NHEAD 4
#define SEQ   4
#define NNODE 20000
#define NEDGE 160000
#define HG    8
#define CG    64
#define NENC  6
#define NDEC  6
#define NMSG  3

typedef __attribute__((ext_vector_type(16))) __bf16 v16bf;
typedef __attribute__((ext_vector_type(8)))  float  v8f;

__device__ __forceinline__ unsigned short f2bf(float x) {
  unsigned u = __float_as_uint(x);
  u += 0x7FFFu + ((u >> 16) & 1u);       // round-to-nearest-even
  return (unsigned short)(u >> 16);
}

// monotonic uint key for float atomicMax (key 0 == "untouched" sentinel)
__device__ __forceinline__ unsigned fkey(float x) {
  unsigned u = __float_as_uint(x);
  return (u & 0x80000000u) ? ~u : (u | 0x80000000u);
}
__device__ __forceinline__ float fdec(unsigned k) {
  if (k == 0u) return 0.0f;
  unsigned u = (k & 0x80000000u) ? (k ^ 0x80000000u) : ~k;
  return __uint_as_float(u);
}

// ---------------------------------------------------------------------------
// WMMA GEMM: out(MxN) = epilogue(A(Mx64) @ W[:, col0:col0+N] + bias)
// 128 threads = 4 waves; wave w owns rows [16*(4*bx+w), +16), block owns a
// 64-wide N strip. N % 64 == 0, M % 16 == 0, K = 64.
// ---------------------------------------------------------------------------
template <bool ACT, bool RES>
__global__ __launch_bounds__(128)
void k_gemm64(const float* __restrict__ A,
              const float* __restrict__ W, int ldw, int col0,
              const float* __restrict__ bias,       // indexed bias[col0+col]
              const float* __restrict__ res,        // residual (MxN) if RES
              float* __restrict__ out,
              int M, int N, float slope) {
  __shared__ unsigned short sA[64 * 64];  // 4 m-tiles x (16x64) bf16
  __shared__ unsigned short sB[64 * 64];  // [n][k] bf16, n in [0,64)

  const int tid  = threadIdx.x;
  const int wave = tid >> 5;
  const int lane = tid & 31;
  const int mt   = blockIdx.x * 4 + wave;      // 16-row tile index
  const int n0   = blockIdx.y * 64;
  const int Mtiles = M >> 4;

  // stage W panel (64k x 64n) as [n][k]; consecutive lanes read consecutive
  // columns of one W row (coalesced), 2 k-rows per sweep.
  for (int i = tid; i < 64 * 64; i += 128) {
    int n = i & 63, k = i >> 6;
    sB[n * 64 + k] = f2bf(W[(size_t)k * ldw + col0 + n0 + n]);
  }
  // each wave stages its own 16x64 A tile with b128 loads
  if (mt < Mtiles) {
    const float4* Ab4 = (const float4*)(A + (size_t)mt * 16 * 64);
    unsigned* sAw = (unsigned*)(sA + wave * 16 * 64);
    for (int i = lane; i < 256; i += 32) {
      float4 f = Ab4[i];
      unsigned lo = (unsigned)f2bf(f.x) | ((unsigned)f2bf(f.y) << 16);
      unsigned hi = (unsigned)f2bf(f.z) | ((unsigned)f2bf(f.w) << 16);
      sAw[i * 2]     = lo;
      sAw[i * 2 + 1] = hi;
    }
  }
  __syncthreads();

  if (mt >= Mtiles) return;                    // wave-uniform: EXEC stays full

  // A fragments per ISA 7.12.2 (16-bit A 16x32, wave32), built once.
  union { v16bf v; unsigned u[8]; } a0, a1;
  const int row = lane & 15;
  const unsigned* arow = (const unsigned*)(sA + ((wave << 4) + row) * 64);
  const int pa = (lane < 16) ? 0 : 4;          // K base / 2
  #pragma unroll
  for (int t = 0; t < 4; ++t) {
    a0.u[t]     = arow[pa + t];                // K = kb + 2t, 2t+1
    a0.u[t + 4] = arow[pa + 8 + t];            // K = kb + 16 + ...
    a1.u[t]     = arow[pa + 16 + t];           // second K-half (32..63)
    a1.u[t + 4] = arow[pa + 24 + t];
  }

  v8f c[4];
  const int pb = (lane < 16) ? 0 : 4;          // lanes 0-15: K 0-7,16-23; 16-31: K 8-15,24-31
  #pragma unroll
  for (int nt = 0; nt < 4; ++nt) {
    union { v16bf v; unsigned u[8]; } b0, b1;
    const unsigned* brow =
        (const unsigned*)(sB + (nt * 16 + (lane & 15)) * 64);
    #pragma unroll
    for (int t = 0; t < 8; ++t) {
      b0.u[t] = brow[pb + t + (t & 4)];
      b1.u[t] = brow[pb + 16 + t + (t & 4)];
    }
    v8f acc = {0.f, 0.f, 0.f, 0.f, 0.f, 0.f, 0.f, 0.f};
    acc = __builtin_amdgcn_wmma_f32_16x16x32_bf16(false, a0.v, false, b0.v,
                                                  (short)0, acc, false, false);
    acc = __builtin_amdgcn_wmma_f32_16x16x32_bf16(false, a1.v, false, b1.v,
                                                  (short)0, acc, false, false);
    c[nt] = acc;
  }

  // Epilogue: lane = N col within subtile, VGPR r -> M = r (+8 upper half)
  const int coll = lane & 15;
  const int mofs = (lane < 16) ? 0 : 8;
  #pragma unroll
  for (int r = 0; r < 8; ++r) {
    const size_t m = (size_t)(mt * 16 + r + mofs);
    #pragma unroll
    for (int nt = 0; nt < 4; ++nt) {
      const int col = n0 + nt * 16 + coll;
      float v = c[nt][r] + bias[col0 + col];
      if (ACT) v = (v >= 0.f) ? v : slope * v;
      if (RES) v += res[m * N + col];
      out[m * N + col] = v;
    }
  }
}

// ---------------------------------------------------------------------------
// LayerNorm over rows of 64: one wave per row, shfl_xor reductions (wave32).
// ---------------------------------------------------------------------------
__global__ __launch_bounds__(256)
void k_ln(const float* __restrict__ in, const float* __restrict__ g,
          const float* __restrict__ b, float* __restrict__ out, int rows) {
  int row  = blockIdx.x * 8 + (threadIdx.x >> 5);
  int lane = threadIdx.x & 31;
  if (row >= rows) return;
  const float* p = in + (size_t)row * 64;
  float x0 = p[lane], x1 = p[lane + 32];
  float s = x0 + x1;
  #pragma unroll
  for (int m = 16; m >= 1; m >>= 1) s += __shfl_xor(s, m, 32);
  float mean = s * (1.f / 64.f);
  float d0 = x0 - mean, d1 = x1 - mean;
  float v2 = d0 * d0 + d1 * d1;
  #pragma unroll
  for (int m = 16; m >= 1; m >>= 1) v2 += __shfl_xor(v2, m, 32);
  float inv = rsqrtf(v2 * (1.f / 64.f) + 1e-5f);
  float* q = out + (size_t)row * 64;
  q[lane]      = d0 * inv * g[lane]      + b[lane];
  q[lane + 32] = d1 * inv * g[lane + 32] + b[lane + 32];
}

// mean over SEQ then LayerNorm (encoder pooling)
__global__ __launch_bounds__(256)
void k_mean_ln(const float* __restrict__ z, const float* __restrict__ g,
               const float* __restrict__ b, float* __restrict__ out) {
  int node = blockIdx.x * 8 + (threadIdx.x >> 5);
  int lane = threadIdx.x & 31;
  if (node >= NNODE) return;
  const size_t S = (size_t)NNODE * 64;
  const float* p = z + (size_t)node * 64;
  float x0 = 0.f, x1 = 0.f;
  #pragma unroll
  for (int s = 0; s < SEQ; ++s) { x0 += p[s * S + lane]; x1 += p[s * S + lane + 32]; }
  x0 *= 0.25f; x1 *= 0.25f;
  float s = x0 + x1;
  #pragma unroll
  for (int m = 16; m >= 1; m >>= 1) s += __shfl_xor(s, m, 32);
  float mean = s * (1.f / 64.f);
  float d0 = x0 - mean, d1 = x1 - mean;
  float v2 = d0 * d0 + d1 * d1;
  #pragma unroll
  for (int m = 16; m >= 1; m >>= 1) v2 += __shfl_xor(v2, m, 32);
  float inv = rsqrtf(v2 * (1.f / 64.f) + 1e-5f);
  float* q = out + (size_t)node * 64;
  q[lane]      = d0 * inv * g[lane]      + b[lane];
  q[lane + 32] = d1 * inv * g[lane + 32] + b[lane + 32];
}

// input projection: z = x(M x 4) @ enc_W(4x64) + enc_b  (K=4, scalar)
__global__ __launch_bounds__(256)
void k_in_proj(const float* __restrict__ x, const float* __restrict__ W,
               const float* __restrict__ b, float* __restrict__ z, int M) {
  int idx = blockIdx.x * 256 + threadIdx.x;
  if (idx >= M * 64) return;
  int m = idx >> 6, c = idx & 63;
  const float* xr = x + (size_t)m * 4;
  float acc = b[c];
  #pragma unroll
  for (int i = 0; i < 4; ++i) acc += xr[i] * W[i * 64 + c];
  z[idx] = acc;
}

// encoder MHA over SEQ=4 keys: one thread per (node, head)
__global__ __launch_bounds__(256)
void k_attn4(const float* __restrict__ qkv, float* __restrict__ o) {
  int idx = blockIdx.x * 256 + threadIdx.x;
  if (idx >= NNODE * NHEAD) return;
  int n = idx >> 2, h = idx & 3;
  const size_t rs = (size_t)NNODE * 192;
  const float* base = qkv + (size_t)n * 192 + h * 16;
  float q[SEQ][16];
  #pragma unroll
  for (int s = 0; s < SEQ; ++s)
    #pragma unroll
    for (int d = 0; d < 16; ++d) q[s][d] = base[s * rs + d];
  float sc[SEQ][SEQ];
  #pragma unroll
  for (int t = 0; t < SEQ; ++t) {
    float kt[16];
    #pragma unroll
    for (int d = 0; d < 16; ++d) kt[d] = base[t * rs + 64 + d];
    #pragma unroll
    for (int s = 0; s < SEQ; ++s) {
      float a = 0.f;
      #pragma unroll
      for (int d = 0; d < 16; ++d) a += q[s][d] * kt[d];
      sc[s][t] = a * 0.25f;                    // 1/sqrt(16)
    }
  }
  #pragma unroll
  for (int s = 0; s < SEQ; ++s) {
    float m = sc[s][0];
    #pragma unroll
    for (int t = 1; t < SEQ; ++t) m = fmaxf(m, sc[s][t]);
    float sum = 0.f;
    #pragma unroll
    for (int t = 0; t < SEQ; ++t) { sc[s][t] = __expf(sc[s][t] - m); sum += sc[s][t]; }
    float inv = 1.f / sum;
    #pragma unroll
    for (int t = 0; t < SEQ; ++t) sc[s][t] *= inv;
  }
  float acc[SEQ][16];
  #pragma unroll
  for (int s = 0; s < SEQ; ++s)
    #pragma unroll
    for (int d = 0; d < 16; ++d) acc[s][d] = 0.f;
  #pragma unroll
  for (int t = 0; t < SEQ; ++t) {
    float vt[16];
    #pragma unroll
    for (int d = 0; d < 16; ++d) vt[d] = base[t * rs + 128 + d];
    #pragma unroll
    for (int s = 0; s < SEQ; ++s)
      #pragma unroll
      for (int d = 0; d < 16; ++d) acc[s][d] += sc[s][t] * vt[d];
  }
  #pragma unroll
  for (int s = 0; s < SEQ; ++s)
    #pragma unroll
    for (int d = 0; d < 16; ++d)
      o[((size_t)s * NNODE + n) * 64 + h * 16 + d] = acc[s][d];
}

// ---------------------------- GAT kernels ----------------------------------
__global__ __launch_bounds__(256)
void k_gat_logit(const float* __restrict__ xl, const float* __restrict__ xr,
                 const float* __restrict__ ea, const float* __restrict__ We,
                 const float* __restrict__ att, const int* __restrict__ src,
                 const int* __restrict__ dst, float* __restrict__ logit,
                 unsigned* __restrict__ lmaxk) {
  int idx = blockIdx.x * 256 + threadIdx.x;
  if (idx >= NEDGE * HG) return;
  int e = idx >> 3, h = idx & 7;
  int s = src[e], d = dst[e];
  const float* pl = xl + (size_t)s * (HG * CG) + h * CG;
  const float* pr = xr + (size_t)d * (HG * CG) + h * CG;
  const float* pw = We + h * CG;
  const float* pa = att + h * CG;
  float a = ea[e], acc = 0.f;
  #pragma unroll 8
  for (int c = 0; c < CG; ++c) {
    float m = pl[c] + pr[c] + a * pw[c];
    m = (m >= 0.f) ? m : 0.2f * m;
    acc += m * pa[c];
  }
  logit[idx] = acc;
  atomicMax(&lmaxk[d * HG + h], fkey(acc));
}

__global__ __launch_bounds__(256)
void k_gat_exp(float* __restrict__ logit, const unsigned* __restrict__ lmaxk,
               const int* __restrict__ dst, float* __restrict__ den) {
  int idx = blockIdx.x * 256 + threadIdx.x;
  if (idx >= NEDGE * HG) return;
  int e = idx >> 3, h = idx & 7;
  int d = dst[e];
  float ex = __expf(logit[idx] - fdec(lmaxk[d * HG + h]));
  logit[idx] = ex;                              // in-place: logit -> ex
  atomicAdd(&den[d * HG + h], ex);
}

__global__ __launch_bounds__(256)
void k_gat_scatter(const float* __restrict__ ex, const float* __restrict__ den,
                   const float* __restrict__ xl, const int* __restrict__ src,
                   const int* __restrict__ dst, float* __restrict__ ydelta) {
  int idx = blockIdx.x * 256 + threadIdx.x;
  if (idx >= NEDGE * CG) return;
  int e = idx >> 6, c = idx & 63;
  int s = src[e], d = dst[e];
  float acc = 0.f;
  #pragma unroll
  for (int h = 0; h < HG; ++h)
    acc += (ex[e * HG + h] / (den[d * HG + h] + 1e-16f)) *
           xl[(size_t)s * (HG * CG) + h * CG + c];
  atomicAdd(&ydelta[(size_t)d * CG + c], 0.125f * acc);   // mean over 8 heads
}

__global__ __launch_bounds__(256)
void k_gat_update(const float* __restrict__ ydelta, const float* __restrict__ bias,
                  float* __restrict__ y) {
  int idx = blockIdx.x * 256 + threadIdx.x;
  if (idx >= NNODE * CG) return;
  float v = ydelta[idx] + bias[idx & 63];
  v = (v >= 0.f) ? v : 0.01f * v;
  y[idx] += v;
}

// final projection: out = yn(N x 64) @ dec_W(64x4) + dec_b
__global__ __launch_bounds__(256)
void k_out_proj(const float* __restrict__ yn, const float* __restrict__ W,
                const float* __restrict__ b, float* __restrict__ out) {
  int idx = blockIdx.x * 256 + threadIdx.x;
  if (idx >= NNODE * 4) return;
  int n = idx >> 2, o = idx & 3;
  float acc = b[o];
  #pragma unroll 8
  for (int j = 0; j < 64; ++j) acc += yn[(size_t)n * 64 + j] * W[j * 4 + o];
  out[idx] = acc;
}

// ---------------------------------------------------------------------------
static void gemm64(hipStream_t st, const float* A, const float* W, int ldw,
                   int col0, const float* bias, const float* res, float* out,
                   int M, int N, bool act, float slope) {
  dim3 g((unsigned)((M / 16 + 3) / 4), (unsigned)(N / 64));
  if (act)
    k_gemm64<true, false><<<g, 128, 0, st>>>(A, W, ldw, col0, bias, nullptr,
                                             out, M, N, slope);
  else if (res)
    k_gemm64<false, true><<<g, 128, 0, st>>>(A, W, ldw, col0, bias, res,
                                             out, M, N, slope);
  else
    k_gemm64<false, false><<<g, 128, 0, st>>>(A, W, ldw, col0, bias, nullptr,
                                              out, M, N, slope);
}

extern "C" void kernel_launch(void* const* d_in, const int* in_sizes, int n_in,
                              void* d_out, int out_size, void* d_ws, size_t ws_size,
                              hipStream_t stream) {
  (void)in_sizes; (void)n_in; (void)out_size; (void)ws_size;
  // pytree leaves in sorted-key DFS order after x, edge_index, edge_attr
  const float* x       = (const float*)d_in[0];
  const int*   ei      = (const int*)  d_in[1];
  const float* eattr   = (const float*)d_in[2];
  const float* dec_W   = (const float*)d_in[3];
  const float* dec_b   = (const float*)d_in[4];
  const float* dW1     = (const float*)d_in[5];
  const float* dW2     = (const float*)d_in[6];
  const float* dWo_c   = (const float*)d_in[7];
  const float* dWo_s   = (const float*)d_in[8];
  const float* dWqkv_c = (const float*)d_in[9];
  const float* dWqkv_s = (const float*)d_in[10];
  const float* db1     = (const float*)d_in[11];
  /* db2 = d_in[12] unused: S=T=1 makes cross-attn independent of q */
  const float* db3     = (const float*)d_in[13];
  const float* dbf1    = (const float*)d_in[14];
  const float* dbf2    = (const float*)d_in[15];
  const float* dbo_c   = (const float*)d_in[16];
  const float* dbo_s   = (const float*)d_in[17];
  const float* dbqkv_c = (const float*)d_in[18];
  const float* dbqkv_s = (const float*)d_in[19];
  const float* dg1     = (const float*)d_in[20];
  const float* dg3     = (const float*)d_in[22];
  const float* enc_W   = (const float*)d_in[23];
  const float* enc_b   = (const float*)d_in[24];
  const float* eW1     = (const float*)d_in[25];
  const float* eW2     = (const float*)d_in[26];
  const float* eWo     = (const float*)d_in[27];
  const float* eWqkv   = (const float*)d_in[28];
  const float* eb1     = (const float*)d_in[29];
  const float* eb2     = (const float*)d_in[30];
  const float* ebf1    = (const float*)d_in[31];
  const float* ebf2    = (const float*)d_in[32];
  const float* ebo     = (const float*)d_in[33];
  const float* ebqkv   = (const float*)d_in[34];
  const float* eg1     = (const float*)d_in[35];
  const float* eg2     = (const float*)d_in[36];
  const float* gWe     = (const float*)d_in[37];
  const float* gWl     = (const float*)d_in[38];
  const float* gWr     = (const float*)d_in[39];
  const float* gatt    = (const float*)d_in[40];
  const float* gbias   = (const float*)d_in[41];
  const float* gbl     = (const float*)d_in[42];
  const float* gbr     = (const float*)d_in[43];
  const float* ndec_b  = (const float*)d_in[44];
  const float* ndec_g  = (const float*)d_in[45];
  const float* nenc_b  = (const float*)d_in[46];
  const float* nenc_g  = (const float*)d_in[47];

  const int M1 = SEQ * NNODE;                       // 80000
  const int* src = ei;
  const int* dst = ei + NEDGE;

  // workspace layout (floats), GAT/decoder phases alias encoder buffers
  float* ws   = (float*)d_ws;
  float* z    = ws;                                 // 5.12M
  float* zn   = ws + 5120000;                       // 5.12M
  float* attn = ws + 10240000;                      // 5.12M
  float* t1   = ws + 15360000;                      // 5.12M
  float* qkv  = ws + 20480000;                      // 15.36M
  float* xr_b   = z;                                // 10.24M (over z+zn)
  float* logitb = attn;                             // 1.28M
  float* vtmp   = attn + 1280000;                   // 1.28M
  float* ydelta = t1;                               // 1.28M
  float* xl_b   = qkv;                              // 10.24M
  float* enc  = ws + 35840000;                      // 1.28M (live thru decoder)
  float* y    = ws + 37120000;
  float* yn   = ws + 38400000;
  unsigned* lmaxk = (unsigned*)(ws + 39680000);     // 160k
  float* den  = ws + 39840000;                      // 160k  (total ~160 MB)

  // ---------------- encoder ----------------
  k_in_proj<<<(M1 * 64 + 255) / 256, 256, 0, stream>>>(x, enc_W, enc_b, z, M1);
  for (int i = 0; i < NENC; ++i) {
    size_t o64 = (size_t)i * 64, o192 = (size_t)i * 192;
    size_t oq = (size_t)i * HID * 192, oo = (size_t)i * HID * HID;
    k_ln<<<(M1 + 7) / 8, 256, 0, stream>>>(z, eg1 + o64, eb1 + o64, zn, M1);
    gemm64(stream, zn, eWqkv + oq, 192, 0, ebqkv + o192, nullptr, qkv, M1, 192, false, 0.f);
    k_attn4<<<(NNODE * NHEAD + 255) / 256, 256, 0, stream>>>(qkv, attn);
    gemm64(stream, attn, eWo + oo, 64, 0, ebo + o64, z, z, M1, 64, false, 0.f);
    k_ln<<<(M1 + 7) / 8, 256, 0, stream>>>(z, eg2 + o64, eb2 + o64, zn, M1);
    gemm64(stream, zn, eW1 + oo, 64, 0, ebf1 + o64, nullptr, t1, M1, 64, true, 0.01f);
    gemm64(stream, t1, eW2 + oo, 64, 0, ebf2 + o64, z, z, M1, 64, false, 0.f);
  }
  k_mean_ln<<<(NNODE + 7) / 8, 256, 0, stream>>>(z, nenc_g, nenc_b, enc);
  hipMemcpyAsync(y, enc, (size_t)NNODE * 64 * sizeof(float),
                 hipMemcpyDeviceToDevice, stream);

  // ---------------- GAT message passing ----------------
  for (int j = 0; j < NMSG; ++j) {
    size_t ow = (size_t)j * HID * (HG * CG), ob = (size_t)j * (HG * CG);
    gemm64(stream, y, gWl + ow, HG * CG, 0, gbl + ob, nullptr, xl_b, NNODE, HG * CG, false, 0.f);
    gemm64(stream, y, gWr + ow, HG * CG, 0, gbr + ob, nullptr, xr_b, NNODE, HG * CG, false, 0.f);
    hipMemsetAsync(lmaxk, 0, (size_t)NNODE * HG * 4, stream);
    hipMemsetAsync(den,   0, (size_t)NNODE * HG * 4, stream);
    hipMemsetAsync(ydelta,0, (size_t)NNODE * CG * 4, stream);
    k_gat_logit<<<(NEDGE * HG + 255) / 256, 256, 0, stream>>>(
        xl_b, xr_b, eattr, gWe + (size_t)j * (HG * CG), gatt + (size_t)j * (HG * CG),
        src, dst, logitb, lmaxk);
    k_gat_exp<<<(NEDGE * HG + 255) / 256, 256, 0, stream>>>(logitb, lmaxk, dst, den);
    k_gat_scatter<<<(NEDGE * CG + 255) / 256, 256, 0, stream>>>(
        logitb, den, xl_b, src, dst, ydelta);
    k_gat_update<<<(NNODE * CG + 255) / 256, 256, 0, stream>>>(
        ydelta, gbias + (size_t)j * CG, y);
  }

  // ---------------- decoder (S=T=1: attention == V,O projections) ----------
  for (int i = 0; i < NDEC; ++i) {
    size_t o64 = (size_t)i * 64, o192 = (size_t)i * 192;
    size_t oq = (size_t)i * HID * 192, oo = (size_t)i * HID * HID;
    // self-attn: y += (LN(y) @ Wv_s + bv_s) @ Wo_s + bo_s
    k_ln<<<(NNODE + 7) / 8, 256, 0, stream>>>(y, dg1 + o64, db1 + o64, yn, NNODE);
    gemm64(stream, yn, dWqkv_s + oq, 192, 128, dbqkv_s + o192, nullptr, vtmp, NNODE, 64, false, 0.f);
    gemm64(stream, vtmp, dWo_s + oo, 64, 0, dbo_s + o64, y, y, NNODE, 64, false, 0.f);
    // cross-attn: y += (enc @ Wv_c + bv_c) @ Wo_c + bo_c   (q is irrelevant)
    gemm64(stream, enc, dWqkv_c + oq, 192, 128, dbqkv_c + o192, nullptr, vtmp, NNODE, 64, false, 0.f);
    gemm64(stream, vtmp, dWo_c + oo, 64, 0, dbo_c + o64, y, y, NNODE, 64, false, 0.f);
    // FFN
    k_ln<<<(NNODE + 7) / 8, 256, 0, stream>>>(y, dg3 + o64, db3 + o64, yn, NNODE);
    gemm64(stream, yn, dW1 + oo, 64, 0, dbf1 + o64, nullptr, t1, NNODE, 64, true, 0.01f);
    gemm64(stream, t1, dW2 + oo, 64, 0, dbf2 + o64, y, y, NNODE, 64, false, 0.f);
  }

  k_ln<<<(NNODE + 7) / 8, 256, 0, stream>>>(y, ndec_g, ndec_b, yn, NNODE);
  k_out_proj<<<(NNODE * 4 + 255) / 256, 256, 0, stream>>>(yn, dec_W, dec_b,
                                                          (float*)d_out);
}
